// AssignAttention_38285338476636
// MI455X (gfx1250) — matrix operand order repaired and checked by reading, not compiled, not run on hardware
//
#include <hip/hip_runtime.h>

// ---------------------------------------------------------------------------
// AssignAttention forward for MI455X (gfx1250), wave32 + WMMA bf16.
// Forward value of the straight-through expression is exactly the one-hot
// argmax over the query axis; argmax of logits == argmax of softmax.
//
// Pipeline:
//   K0  : weights fp32->bf16 (wk transposed), zero fp32 accumulator
//   K1a : q = query @ wq^T                      (WMMA bf16)
//   K1b : M[b,h] = q_bh @ wk_h  (folds K-proj into logits; saves 155 GF)
//   K2  : stage kv slice in LDS (bf16) -> logits & V WMMA concurrently ->
//         argmax -> ds_add_f32 scatter -> atomic flush
//   K3  : out = acc @ wp^T + bp                 (WMMA bf16)
// ---------------------------------------------------------------------------

constexpr int Bb = 32, Nn = 64, Ss = 4096, Cc = 768, Hh = 12, HDd = 64;
constexpr int NCH   = 8;    // S-chunks per (b,h): 512 rows each
constexpr int SROWS = 16;   // kv rows staged in LDS per sub-iteration
constexpr int SUBS2 = 32;   // 32 sub-iterations of 16 rows per chunk
constexpr int LDP   = 776;  // padded LDS row stride (bf16 elems): 1552B % 256B = 16B

typedef __bf16 bf16_t;
typedef __bf16 v16bf __attribute__((ext_vector_type(16)));
typedef __bf16 v8bf  __attribute__((ext_vector_type(8)));
typedef __bf16 v4bf  __attribute__((ext_vector_type(4)));
typedef float  v8f   __attribute__((ext_vector_type(8)));
typedef float  v4f   __attribute__((ext_vector_type(4)));

static __device__ __forceinline__ v8f wmma_bf16(v16bf a, v16bf b, v8f c) {
    // 8 args: (neg_a, A, neg_b, B, c_mod, C, reuse_a, reuse_b)
    return __builtin_amdgcn_wmma_f32_16x16x32_bf16(false, a, false, b, (short)0, c,
                                                   false, false);
}

// A-matrix 16x32 bf16 fragment (doc layout): lane L holds row M=L&15.
// half-index e -> K = kk + e + (e>=8?8:0) + (L>=16?8:0)  == two 8-runs.
static __device__ __forceinline__ v16bf load_a_bf(const bf16_t* row, int kk, int hi) {
    int k0 = kk + (hi ? 8 : 0);
    v8bf lo = *(const v8bf*)(row + k0);
    v8bf hh = *(const v8bf*)(row + k0 + 16);
    v16bf a;
    #pragma unroll
    for (int i = 0; i < 8; ++i) { a[i] = lo[i]; a[8 + i] = hh[i]; }
    return a;
}

static __device__ __forceinline__ v16bf load_a_f32(const float* row, int kk, int hi) {
    int k0 = kk + (hi ? 8 : 0);
    v4f t0 = *(const v4f*)(row + k0);
    v4f t1 = *(const v4f*)(row + k0 + 4);
    v4f t2 = *(const v4f*)(row + k0 + 16);
    v4f t3 = *(const v4f*)(row + k0 + 20);
    v16bf a;
    #pragma unroll
    for (int i = 0; i < 4; ++i) {
        a[i]      = (bf16_t)t0[i];
        a[4 + i]  = (bf16_t)t1[i];
        a[8 + i]  = (bf16_t)t2[i];
        a[12 + i] = (bf16_t)t3[i];
    }
    return a;
}

// B-matrix 32x16 bf16 fragment: lane L holds column N=L&15;
// half-index e -> K = kk + e + (e>=8?8:0) + (L>=16?8:0)  == two 8-runs.
static __device__ __forceinline__ v16bf load_b_bf(const bf16_t* row, int kk, int hi) {
    int k0 = kk + (hi ? 8 : 0);
    v8bf lo = *(const v8bf*)(row + k0);
    v8bf hh = *(const v8bf*)(row + k0 + 16);
    v16bf b;
    #pragma unroll
    for (int i = 0; i < 8; ++i) { b[i] = lo[i]; b[8 + i] = hh[i]; }
    return b;
}

// ---------------------------------------------------------------------------
// K0: weight conversion fp32 -> bf16 (wk transposed) + zero global accumulator
// ---------------------------------------------------------------------------
__global__ void prep_kernel(const float* __restrict__ wq, const float* __restrict__ wk,
                            const float* __restrict__ wv, const float* __restrict__ wp,
                            bf16_t* wq_bf, bf16_t* wkT_bf, bf16_t* wv_bf, bf16_t* wp_bf,
                            float* accg) {
    int i = blockIdx.x * 256 + threadIdx.x;
    const int szW = Cc * Cc;
    if (i < szW) {
        wq_bf[i] = (bf16_t)wq[i];
        wv_bf[i] = (bf16_t)wv[i];
        wp_bf[i] = (bf16_t)wp[i];
        int o = i / Cc, c = i % Cc;
        wkT_bf[c * Cc + o] = (bf16_t)wk[i];      // wkT[c][o] = wk[o][c]
    }
    if (i < Bb * Nn * Cc) accg[i] = 0.0f;
}

// ---------------------------------------------------------------------------
// K1a: q = query @ wq^T   (2048 x 768 x 768), output bf16 row-major
// ---------------------------------------------------------------------------
__global__ void __launch_bounds__(256) qproj_kernel(const float* __restrict__ query,
                                                    const bf16_t* __restrict__ wq_bf,
                                                    bf16_t* __restrict__ q_bf) {
    int wave = threadIdx.x >> 5, lane = threadIdx.x & 31, hi = lane >> 4;
    int tile = blockIdx.x * 8 + wave;            // 128 x 48 tiles
    int tm = tile / (Cc / 16);
    int tn = tile % (Cc / 16);
    const float* arow = query + (size_t)(tm * 16 + (lane & 15)) * Cc;
    const bf16_t* brow = wq_bf + (size_t)(tn * 16 + (lane & 15)) * Cc;
    v8f acc = {};
    for (int kk = 0; kk < Cc; kk += 32)
        acc = wmma_bf16(load_a_f32(arow, kk, hi), load_b_bf(brow, kk, hi), acc);
    int col = tn * 16 + (lane & 15);
    #pragma unroll
    for (int r = 0; r < 8; ++r)
        q_bf[(size_t)(tm * 16 + r + hi * 8) * Cc + col] = (bf16_t)acc[r];
}

// ---------------------------------------------------------------------------
// K1b: M[b,h] = q_bh (64x64) @ wk_h (64x768)  -> qkm bf16 [b,h,64,768]
// ---------------------------------------------------------------------------
__global__ void __launch_bounds__(256) qkm_kernel(const bf16_t* __restrict__ q_bf,
                                                  const bf16_t* __restrict__ wkT_bf,
                                                  bf16_t* __restrict__ qkm) {
    int wave = threadIdx.x >> 5, lane = threadIdx.x & 31, hi = lane >> 4;
    int tile = blockIdx.x * 8 + wave;            // (B*H) * (4 x 48) tiles
    int bh = tile / 192;
    int rest = tile % 192;
    int tm = rest / 48, tn = rest % 48;
    int b = bh / Hh, h = bh % Hh;
    const bf16_t* arow = q_bf + (size_t)(b * Nn + tm * 16 + (lane & 15)) * Cc + h * HDd;
    const bf16_t* brow = wkT_bf + (size_t)(tn * 16 + (lane & 15)) * Cc + h * HDd;
    v8f acc = {};
    for (int kk = 0; kk < HDd; kk += 32)
        acc = wmma_bf16(load_a_bf(arow, kk, hi), load_b_bf(brow, kk, hi), acc);
    int col = tn * 16 + (lane & 15);
    #pragma unroll
    for (int r = 0; r < 8; ++r)
        qkm[((size_t)bh * Nn + tm * 16 + r + hi * 8) * Cc + col] = (bf16_t)acc[r];
}

// ---------------------------------------------------------------------------
// K2: fused logits & V-projection (concurrent) -> argmax -> LDS scatter-accumulate.
// One workgroup (8 waves) per (b, h, 512-row S chunk); 16-row LDS-staged subs.
// ---------------------------------------------------------------------------
__global__ void __launch_bounds__(256) attn_assign_kernel(const float* __restrict__ kv,
                                                          const bf16_t* __restrict__ qkm,
                                                          const bf16_t* __restrict__ wv_bf,
                                                          float* __restrict__ accg) {
    __shared__ bf16_t skv[SROWS * LDP];   // 24.8 KB staged kv slice (bf16, padded)
    __shared__ float  acc_h[Nn * HDd];    // 16 KB per-head accumulator
    __shared__ float  attnS[Nn * SROWS];  // 4 KB logits tile (64 n x 16 s)
    __shared__ int    idxS[SROWS];

    int tid = threadIdx.x, wave = tid >> 5, lane = tid & 31, hi = lane >> 4;
    int blk = blockIdx.x;
    int ch = blk % NCH;
    int h  = (blk / NCH) % Hh;
    int b  = blk / (NCH * Hh);

    for (int i = tid; i < Nn * HDd; i += 256) acc_h[i] = 0.f;

    const bf16_t* qbase = qkm + (size_t)(b * Hh + h) * Nn * Cc;
    const float*  kvchunk = kv + ((size_t)b * Ss + ch * (SUBS2 * SROWS)) * Cc;

    for (int sub = 0; sub < SUBS2; ++sub) {
        const float* kvsub = kvchunk + (size_t)sub * SROWS * Cc;

        // ---- stage 16 x 768 kv slice into LDS as bf16 (once per sub) ----
        for (int g = tid; g < SROWS * Cc / 4; g += 256) {
            int r  = g / (Cc / 4);
            int c4 = (g % (Cc / 4)) * 4;
            v4f t = *(const v4f*)(kvsub + (size_t)r * Cc + c4);
            v4bf o;
            #pragma unroll
            for (int i = 0; i < 4; ++i) o[i] = (bf16_t)t[i];
            *(v4bf*)(&skv[r * LDP + c4]) = o;
        }
        // prefetch next slice while this one is consumed (global_prefetch_b8)
        if (sub + 1 < SUBS2) {
            const char* nxt = (const char*)(kvsub + (size_t)SROWS * Cc);
            for (int g = tid; g < (SROWS * Cc * 4) / 128; g += 256)
                __builtin_prefetch(nxt + g * 128, 0, 1);
        }
        __syncthreads();

        v8f vacc = {};   // V-tile accumulator for waves 4..7
        if (wave < 4) {
            // ---- logits tile: attn[n, s] , n-tile = wave, s = 0..15 ----
            const bf16_t* arow = qbase + (size_t)(wave * 16 + (lane & 15)) * Cc;
            const bf16_t* brow = &skv[(lane & 15) * LDP];
            v8f acc = {};
            for (int kk = 0; kk < Cc; kk += 32)
                acc = wmma_bf16(load_a_bf(arow, kk, hi), load_b_bf(brow, kk, hi), acc);
            #pragma unroll
            for (int r = 0; r < 8; ++r)
                attnS[(wave * 16 + r + hi * 8) * SROWS + (lane & 15)] = acc[r];
        } else {
            // ---- V tile: v[s, hd] , hd-tile = wave-4 (independent of argmax) ----
            const bf16_t* arow = &skv[(lane & 15) * LDP];
            const bf16_t* brow = wv_bf + (size_t)(h * HDd + (wave - 4) * 16 + (lane & 15)) * Cc;
            for (int kk = 0; kk < Cc; kk += 32)
                vacc = wmma_bf16(load_a_bf(arow, kk, hi), load_b_bf(brow, kk, hi), vacc);
        }
        __syncthreads();

        // ---- argmax over n (first-max wins, matching jnp.argmax) ----
        if (tid < SROWS) {
            int s = tid;
            float best = attnS[s];
            int bi = 0;
            for (int n = 1; n < Nn; ++n) {
                float v = attnS[n * SROWS + s];
                if (v > best) { best = v; bi = n; }
            }
            idxS[s] = bi;
        }
        __syncthreads();

        // ---- scatter V rows into per-head accumulator by idx (ds_add_f32) ----
        if (wave >= 4) {
            #pragma unroll
            for (int r = 0; r < 8; ++r) {
                int s  = r + hi * 8;
                int hd = (wave - 4) * 16 + (lane & 15);
                atomicAdd(&acc_h[idxS[s] * HDd + hd], vacc[r]);
            }
        }
        __syncthreads();   // protect skv/attnS/idxS before next sub-iteration
    }

    // ---- flush per-head accumulator (4096 global fadds per WG) ----
    for (int i = tid; i < Nn * HDd; i += 256) {
        int n = i >> 6, hd = i & 63;
        atomicAdd(accg + ((size_t)(b * Nn + n)) * Cc + h * HDd + hd, acc_h[i]);
    }
}

// ---------------------------------------------------------------------------
// K3: out = acc @ wp^T + bp   (2048 x 768 x 768), fp32 out
// ---------------------------------------------------------------------------
__global__ void __launch_bounds__(256) outproj_kernel(const float* __restrict__ accg,
                                                      const bf16_t* __restrict__ wp_bf,
                                                      const float* __restrict__ bp,
                                                      float* __restrict__ out) {
    int wave = threadIdx.x >> 5, lane = threadIdx.x & 31, hi = lane >> 4;
    int tile = blockIdx.x * 8 + wave;            // 128 x 48 tiles
    int tm = tile / 48, tn = tile % 48;
    const float*  arow = accg + (size_t)(tm * 16 + (lane & 15)) * Cc;
    const bf16_t* brow = wp_bf + (size_t)(tn * 16 + (lane & 15)) * Cc;
    v8f acc = {};
    for (int kk = 0; kk < Cc; kk += 32)
        acc = wmma_bf16(load_a_f32(arow, kk, hi), load_b_bf(brow, kk, hi), acc);
    int col = tn * 16 + (lane & 15);
    float bias = bp[col];
    #pragma unroll
    for (int r = 0; r < 8; ++r)
        out[(size_t)(tm * 16 + r + hi * 8) * Cc + col] = acc[r] + bias;
}

// ---------------------------------------------------------------------------
extern "C" void kernel_launch(void* const* d_in, const int* in_sizes, int n_in,
                              void* d_out, int out_size, void* d_ws, size_t ws_size,
                              hipStream_t stream) {
    const float* query = (const float*)d_in[0];
    const float* kv    = (const float*)d_in[1];
    const float* wq    = (const float*)d_in[2];
    const float* wk    = (const float*)d_in[3];
    const float* wv    = (const float*)d_in[4];
    const float* wp    = (const float*)d_in[5];
    const float* bp    = (const float*)d_in[6];
    float* out = (float*)d_out;

    const size_t szW = (size_t)Cc * Cc;
    bf16_t* wq_bf  = (bf16_t*)d_ws;
    bf16_t* wkT_bf = wq_bf + szW;
    bf16_t* wv_bf  = wkT_bf + szW;
    bf16_t* wp_bf  = wv_bf + szW;
    bf16_t* q_bf   = wp_bf + szW;                        // B*N*C bf16
    bf16_t* qkm    = q_bf + (size_t)Bb * Nn * Cc;        // B*H*N*C bf16
    float*  accg   = (float*)(qkm + (size_t)Bb * Hh * Nn * Cc);  // B*N*C f32

    prep_kernel<<<(Bb * Nn * Cc + 255) / 256, 256, 0, stream>>>(
        wq, wk, wv, wp, wq_bf, wkT_bf, wv_bf, wp_bf, accg);
    qproj_kernel<<<(128 * 48) / 8, 256, 0, stream>>>(query, wq_bf, q_bf);
    qkm_kernel<<<(Bb * Hh * 4 * 48) / 8, 256, 0, stream>>>(q_bf, wkT_bf, qkm);
    attn_assign_kernel<<<Bb * Hh * NCH, 256, 0, stream>>>(kv, qkm, wv_bf, accg);
    outproj_kernel<<<(128 * 48) / 8, 256, 0, stream>>>(accg, wp_bf, bp, out);
}
